// KLattention_27745488733033
// MI455X (gfx1250) — matrix operand order repaired and checked by reading, not compiled, not run on hardware
//
#include <hip/hip_runtime.h>
#include <hip/hip_bf16.h>

// ---------------------------------------------------------------------------
// KL-gated windowed attention for MI455X (gfx1250, wave32, WMMA).
//   1) lse_kernel        : per-(b,seg,d) logsumexp over the 256-row window
//   2) kl_partial_kernel : per-(boundary,b) partial KL sums (deterministic)
//   3) kl_top3_kernel    : reduce over b, stable 3-smallest selection
//   4) cvt_T/zero/build  : f32->f16 transposed weights, zero V^T pad, gather h
//   5) gemm_kernel x3    : Q,K,V projections; A-tile double-buffered via
//                          global_load_async_to_lds_b128 (ASYNCcnt) overlapped
//                          with WMMA compute; B frags read contiguously from
//                          pre-transposed W^T (L2-resident)
//   6) attn_kernel       : flash attention, 1 wave = (seq, head, 16-query tile)
//   7) gemm_kernel       : O @ Wo + bo with scatter epilogue into d_out
// ---------------------------------------------------------------------------

typedef _Float16 f16_t;
typedef __attribute__((ext_vector_type(16))) _Float16 v16h;
typedef __attribute__((ext_vector_type(8)))  float    v8f;
typedef __attribute__((ext_vector_type(4)))  float    v4f;

union FragU {
    v16h v;
    v4f  f2[2];
    f16_t e[16];
};

// Problem constants
#define NB_B    8
#define SEQ     8192
#define DMODEL  512
#define WINL    256
#define STRIDEL 128
#define NSEG    63
#define NBOUND  62
#define TOPK    3
#define LSEQ    513           // 1 + 2*WIN
#define NBATCH  24            // TOPK * B
#define MTOT    12312         // NBATCH * LSEQ
#define NHEAD   8
#define HD      64
#define LTPAD   544           // LSEQ padded to 17*32 for V^T rows
#define NKT     17            // key tiles of 32
#define NQT     33            // query tiles of 16
#define NTASKS  (NBATCH * NHEAD * NQT)   // 6336 wave-tasks
#define KSTEPS  (DMODEL / 32) // 16

// ---------------------------------------------------------------------------
// CDNA5 async global->LDS copy (ASYNCcnt) - CDNA5 ISA
// GVS mode: mem_addr = SADDR + VADDR(i32) ; LDS dest addr from VDST vgpr.
// ---------------------------------------------------------------------------
__device__ __forceinline__ void async_load_b128(uint32_t lds_off, const void* sbase,
                                                uint32_t voff) {
    asm volatile("global_load_async_to_lds_b128 %0, %1, %2"
                 :: "v"(lds_off), "v"(voff), "s"(sbase)
                 : "memory");
}
__device__ __forceinline__ void wait_asynccnt0() {
    asm volatile("s_wait_asynccnt 0" ::: "memory");
}

// ---------------------------------------------------------------------------
// WMMA fragment loaders (CDNA5 ISA §7.12.2, wave32)
// A 16x32 f16: lane m=lane&15, hi=lane>>4; elems 0..15 = K hi*16..hi*16+15
//              (two 16B chunks)
// B 32x16 f16 from N-major storage: lane n=lane&15, kh=lane>>4;
//              elems 0..15 = K kh*16..kh*16+15 (one contiguous 32B run)
// C/D 16x16 f32: vgpr r -> row r + 8*(lane>>4), col = lane&15
// ---------------------------------------------------------------------------
__device__ __forceinline__ v16h load_frag_a(const f16_t* base, int stride, int lane) {
    int m = lane & 15, hi = lane >> 4;
    const f16_t* p = base + (size_t)m * stride;
    FragU u;
    u.f2[0] = *(const v4f*)(p + hi * 16);
    u.f2[1] = *(const v4f*)(p + hi * 16 + 8);
    return u.v;
}

// ---------------------------------------------------------------------------
// Stage 1: logsumexp over each window column (online, one pass)
// ---------------------------------------------------------------------------
__global__ __launch_bounds__(256) void lse_kernel(const float* __restrict__ x,
                                                  float* __restrict__ lse) {
    int blk = blockIdx.x;
    int s = blk % NSEG, b = blk / NSEG;
    const float* base = x + ((size_t)b * SEQ + (size_t)s * STRIDEL) * DMODEL;
    for (int d = threadIdx.x; d < DMODEL; d += 256) {
        float m = -INFINITY, sum = 0.f;
        for (int t = 0; t < WINL; ++t) {
            float v = base[(size_t)t * DMODEL + d];
            float mn = fmaxf(m, v);
            sum = sum * __expf(m - mn) + __expf(v - mn);
            m = mn;
        }
        lse[((size_t)b * NSEG + s) * DMODEL + d] = m + __logf(sum);
    }
}

// ---------------------------------------------------------------------------
// Stage 2: KL partials per (boundary,b):
//   sum_d [ sum_t exp(x1-lse1)*(x1-x0) - (lse1-lse0) ]
// ---------------------------------------------------------------------------
__global__ __launch_bounds__(256) void kl_partial_kernel(const float* __restrict__ x,
                                                         const float* __restrict__ lse,
                                                         float* __restrict__ partial) {
    int blk = blockIdx.x;
    int s = blk % NBOUND, b = blk / NBOUND;
    const float* x0 = x + ((size_t)b * SEQ + (size_t)s * STRIDEL) * DMODEL;
    const float* x1 = x0 + (size_t)STRIDEL * DMODEL;
    float acc = 0.f;
    for (int d = threadIdx.x; d < DMODEL; d += 256) {
        float l0 = lse[((size_t)b * NSEG + s) * DMODEL + d];
        float l1 = lse[((size_t)b * NSEG + s + 1) * DMODEL + d];
        float a = 0.f;
        for (int t = 0; t < WINL; ++t) {
            float v1 = x1[(size_t)t * DMODEL + d];
            float v0 = x0[(size_t)t * DMODEL + d];
            a += __expf(v1 - l1) * (v1 - v0);
        }
        acc += a - (l1 - l0);
    }
    __shared__ float red[256];
    red[threadIdx.x] = acc;
    __syncthreads();
    for (int w = 128; w > 0; w >>= 1) {
        if (threadIdx.x < (unsigned)w) red[threadIdx.x] += red[threadIdx.x + w];
        __syncthreads();
    }
    if (threadIdx.x == 0) partial[s * NB_B + b] = red[0];
}

// ---------------------------------------------------------------------------
// Stage 3: reduce over b (fixed order), stable 3-min selection
// ---------------------------------------------------------------------------
__global__ void kl_top3_kernel(const float* __restrict__ partial, int* __restrict__ top3) {
    __shared__ float kl[NBOUND];
    int s = threadIdx.x;
    if (s < NBOUND) {
        float v = 0.f;
        for (int b = 0; b < NB_B; ++b) v += partial[s * NB_B + b];
        kl[s] = v;
    }
    __syncthreads();
    if (threadIdx.x == 0) {
        for (int j = 0; j < TOPK; ++j) {
            int   bi = 0;
            float bv = INFINITY;
            for (int i = 0; i < NBOUND; ++i) {
                if (kl[i] < bv) { bv = kl[i]; bi = i; }   // strict < == stable argsort
            }
            top3[j] = bi;
            kl[bi] = INFINITY;
        }
    }
}

// ---------------------------------------------------------------------------
// Utility: f32 (K-major) -> f16 transposed (N-major) weight convert; f16 zero
// ---------------------------------------------------------------------------
__global__ void cvt_f16_T_kernel(const float* __restrict__ src, f16_t* __restrict__ dst) {
    int i = blockIdx.x * 256 + threadIdx.x;   // i over K*N
    int k = i >> 9, n = i & 511;
    dst[(size_t)n * DMODEL + k] = (f16_t)src[i];
}

__global__ void zero_f16_kernel(f16_t* __restrict__ p, int n) {
    int i = blockIdx.x * 256 + threadIdx.x;
    if (i < n) p[i] = (f16_t)0.f;
}

// ---------------------------------------------------------------------------
// Stage 4: gather h = [cls | segA | segB] as f16, layout (TOPK*B, 513, 512)
// ---------------------------------------------------------------------------
__global__ __launch_bounds__(256) void build_h_kernel(const float* __restrict__ x,
                                                      const float* __restrict__ klcls,
                                                      const int* __restrict__ top3,
                                                      f16_t* __restrict__ h) {
    int row = blockIdx.x;            // 0 .. MTOT-1
    int l = row % LSEQ;
    int nb = row / LSEQ;
    int b = nb % NB_B;
    int kk = nb / NB_B;
    const float* src;
    if (l == 0) {
        src = klcls + (size_t)b * DMODEL;
    } else {
        int st = top3[kk] * STRIDEL;
        int t = (l <= WINL) ? (st + l - 1) : (st + l - 129);   // segB start = st+128
        src = x + ((size_t)b * SEQ + t) * DMODEL;
    }
    f16_t* dst = h + (size_t)row * DMODEL;
    for (int d = threadIdx.x; d < DMODEL; d += 256) dst[d] = (f16_t)src[d];
}

// ---------------------------------------------------------------------------
// Stage 5/7: tiled WMMA GEMM  C(MTOT x 512) = A(MTOT x 512) @ W(512 x 512)
// Block tile 128x64, 8 waves (4Mx2N), wave tile 32x32, K-step 32.
// A tile DOUBLE-BUFFERED in LDS via global_load_async_to_lds_b128: the async
// copy of tile i+1 overlaps WMMA compute on tile i (one barrier per K-step).
// B frags read directly from pre-transposed (N-major) W^T in global.
// epilogue 0: f16 out (scaled, +bias)            (Q, K)
// epilogue 1: f16 out + transposed V^T store     (V)
// epilogue 2: f32 scatter into d_out             (O-proj)
// ---------------------------------------------------------------------------
__global__ __launch_bounds__(256) void gemm_kernel(const f16_t* __restrict__ A,
                                                   const f16_t* __restrict__ Wt,
                                                   const float* __restrict__ bias,
                                                   float scale, int epilogue,
                                                   f16_t* __restrict__ out16,
                                                   f16_t* __restrict__ vt,
                                                   float* __restrict__ out32) {
    __shared__ __attribute__((aligned(16))) f16_t As[2][128 * 32];

    int tid = threadIdx.x;
    int lane = tid & 31, widx = tid >> 5;
    int wm = widx & 3, wn = widx >> 2;
    int m0 = blockIdx.y * 128, n0 = blockIdx.x * 64;
    int hi = lane >> 4, ln = lane & 15;

    v8f c00 = {}, c01 = {}, c10 = {}, c11 = {};

    // async staging: 128 rows x 32 halfs = 512 x 16B chunks, 2 per thread.
    // Tail rows are clamped (duplicate data; never stored).
    int ch0 = tid * 2, ch1 = tid * 2 + 1;
    int r0 = ch0 >> 2, s0 = ch0 & 3;
    int r1 = ch1 >> 2, s1 = ch1 & 3;
    int gm0 = m0 + r0; if (gm0 > MTOT - 1) gm0 = MTOT - 1;
    int gm1 = m0 + r1; if (gm1 > MTOT - 1) gm1 = MTOT - 1;
    uint32_t ldsA[2][2];
#pragma unroll
    for (int bf = 0; bf < 2; ++bf) {
        ldsA[bf][0] = (uint32_t)(uintptr_t)(&As[bf][0] + r0 * 32 + s0 * 8);
        ldsA[bf][1] = (uint32_t)(uintptr_t)(&As[bf][0] + r1 * 32 + s1 * 8);
    }
    uint32_t goA = (uint32_t)(((size_t)gm0 * DMODEL + s0 * 8) * sizeof(f16_t));
    uint32_t goB = (uint32_t)(((size_t)gm1 * DMODEL + s1 * 8) * sizeof(f16_t));

    const f16_t* wb0 = Wt + (size_t)(n0 + wn * 32 + ln) * DMODEL;        // N-subtile 0
    const f16_t* wb1 = Wt + (size_t)(n0 + wn * 32 + 16 + ln) * DMODEL;   // N-subtile 1

    // prologue: stage K-step 0 into buffer 0
    async_load_b128(ldsA[0][0], A, goA);
    async_load_b128(ldsA[0][1], A, goB);

    for (int it = 0; it < KSTEPS; ++it) {
        int kk = it * 32;
        wait_asynccnt0();        // this wave's async copies (tile it) landed
        __syncthreads();         // everyone's landed; prev tile fully consumed

        if (it + 1 < KSTEPS) {   // overlap: stage tile it+1 into other buffer
            uint32_t koff = (uint32_t)((it + 1) * 32 * sizeof(f16_t));
            async_load_b128(ldsA[(it + 1) & 1][0], A, goA + koff);
            async_load_b128(ldsA[(it + 1) & 1][1], A, goB + koff);
        }

        const f16_t* Ab = &As[it & 1][0];
        v16h a0 = load_frag_a(Ab + (size_t)(wm * 32) * 32, 32, lane);
        v16h a1 = load_frag_a(Ab + (size_t)(wm * 32 + 16) * 32, 32, lane);

        FragU b0, b1;
        b0.f2[0] = *(const v4f*)(wb0 + kk + hi * 16);
        b0.f2[1] = *(const v4f*)(wb0 + kk + hi * 16 + 8);
        b1.f2[0] = *(const v4f*)(wb1 + kk + hi * 16);
        b1.f2[1] = *(const v4f*)(wb1 + kk + hi * 16 + 8);

        c00 = __builtin_amdgcn_wmma_f32_16x16x32_f16(false, a0, false, b0.v, (short)0, c00, false, false);
        c01 = __builtin_amdgcn_wmma_f32_16x16x32_f16(false, a0, false, b1.v, (short)0, c01, false, false);
        c10 = __builtin_amdgcn_wmma_f32_16x16x32_f16(false, a1, false, b0.v, (short)0, c10, false, false);
        c11 = __builtin_amdgcn_wmma_f32_16x16x32_f16(false, a1, false, b1.v, (short)0, c11, false, false);
    }

    v8f cs0[2] = {c00, c01};
    v8f cs1[2] = {c10, c11};
#pragma unroll
    for (int i = 0; i < 2; ++i) {
#pragma unroll
        for (int j = 0; j < 2; ++j) {
            v8f c = i ? cs1[j] : cs0[j];
            int gn = n0 + wn * 32 + j * 16 + ln;
            float bv = bias[gn];
#pragma unroll
            for (int r = 0; r < 8; ++r) {
                int gm = m0 + wm * 32 + i * 16 + r + 8 * hi;
                if (gm >= MTOT) continue;
                float val = (c[r] + bv) * scale;
                if (epilogue == 2) {
                    int l = gm % LSEQ, nb = gm / LSEQ;
                    int b = nb & 7, k3 = nb >> 3;
                    size_t orow = (l == 0) ? (size_t)(12288 + b * TOPK + k3)
                                           : (size_t)(b * 1536 + k3 * 512 + (l - 1));
                    out32[orow * DMODEL + gn] = val;
                } else {
                    out16[(size_t)gm * DMODEL + gn] = (f16_t)val;
                    if (epilogue == 1) {
                        int head = gn >> 6, dim = gn & 63;
                        int nb = gm / LSEQ, l = gm % LSEQ;
                        vt[((size_t)(nb * NHEAD + head) * HD + dim) * LTPAD + l] = (f16_t)val;
                    }
                }
            }
        }
    }
}

// ---------------------------------------------------------------------------
// Stage 6: flash attention. One wave = (seq n, head, 16-query tile).
// ---------------------------------------------------------------------------
__global__ __launch_bounds__(256) void attn_kernel(const f16_t* __restrict__ q,
                                                   const f16_t* __restrict__ kmat,
                                                   const f16_t* __restrict__ vt,
                                                   f16_t* __restrict__ o) {
    __shared__ __attribute__((aligned(16))) f16_t pbuf[8][16 * 32];

    int lane = threadIdx.x & 31, widx = threadIdx.x >> 5;
    int task = blockIdx.x * 8 + widx;
    if (task >= NTASKS) return;
    int qt = task % NQT;
    int rem = task / NQT;
    int head = rem & 7;
    int n = rem >> 3;
    int hc = head * HD;
    int hi = lane >> 4, ln = lane & 15;

    // Q A-fragments (K = 64 -> two 16x32 chunks); invalid query rows -> 0
    int qrow = qt * 16 + ln;
    bool qv = qrow < LSEQ;
    const f16_t* qp = q + ((size_t)(n * LSEQ + (qv ? qrow : 0))) * DMODEL + hc;
    v16h aq[2];
#pragma unroll
    for (int kc = 0; kc < 2; ++kc) {
        FragU u;
        u.f2[0] = *(const v4f*)(qp + kc * 32 + hi * 16);
        u.f2[1] = *(const v4f*)(qp + kc * 32 + hi * 16 + 8);
        if (!qv) { u.f2[0] = (v4f){}; u.f2[1] = (v4f){}; }
        aq[kc] = u.v;
    }

    v8f oa[4] = {};
    float mrow[8], lrow[8];
#pragma unroll
    for (int r = 0; r < 8; ++r) { mrow[r] = -1e30f; lrow[r] = 0.f; }

    f16_t* pw = &pbuf[widx][0];
    const f16_t* vbase = vt + (size_t)(n * NHEAD + head) * HD * LTPAD;

    for (int kt = 0; kt < NKT; ++kt) {
        int kb = kt * 32;
        v8f s[2] = {};
#pragma unroll
        for (int ns = 0; ns < 2; ++ns) {
            int key = kb + ns * 16 + ln;
            int keyc = key < LSEQ ? key : (LSEQ - 1);
            const f16_t* kp = kmat + ((size_t)(n * LSEQ + keyc)) * DMODEL + hc;
#pragma unroll
            for (int kc = 0; kc < 2; ++kc) {
                FragU u;   // B frag: contiguous 16 dims of this key's row
                u.f2[0] = *(const v4f*)(kp + kc * 32 + hi * 16);
                u.f2[1] = *(const v4f*)(kp + kc * 32 + hi * 16 + 8);
                s[ns] = __builtin_amdgcn_wmma_f32_16x16x32_f16(false, aq[kc], false, u.v,
                                                               (short)0, s[ns], false, false);
            }
            if (key >= LSEQ) {
#pragma unroll
                for (int r = 0; r < 8; ++r) s[ns][r] = -1e30f;
            }
        }

        // flash softmax update (row stats live per-vgpr-slot, half-wave reduce)
#pragma unroll
        for (int r = 0; r < 8; ++r) {
            float vmax = fmaxf(s[0][r], s[1][r]);
#pragma unroll
            for (int off = 1; off < 16; off <<= 1)
                vmax = fmaxf(vmax, __shfl_xor(vmax, off, 16));
            float mn = fmaxf(mrow[r], vmax);
            float alpha = __expf(mrow[r] - mn);
            mrow[r] = mn;
            float p0 = __expf(s[0][r] - mn);
            float p1 = __expf(s[1][r] - mn);
            float rs = p0 + p1;
#pragma unroll
            for (int off = 1; off < 16; off <<= 1)
                rs += __shfl_xor(rs, off, 16);
            lrow[r] = lrow[r] * alpha + rs;
#pragma unroll
            for (int j = 0; j < 4; ++j) oa[j][r] *= alpha;
            int m = r + 8 * hi;
            pw[m * 32 + ln] = (f16_t)p0;
            pw[m * 32 + 16 + ln] = (f16_t)p1;
        }

        // cross-lane LDS RAW inside the wave: wait DS ops, then re-fragment
        asm volatile("s_wait_dscnt 0" ::: "memory");
        v16h ap = load_frag_a(pw, 32, lane);

#pragma unroll
        for (int ds = 0; ds < 4; ++ds) {
            const f16_t* vp = vbase + (size_t)(ds * 16 + ln) * LTPAD + kb + hi * 16;
            FragU u;
            u.f2[0] = *(const v4f*)(vp);
            u.f2[1] = *(const v4f*)(vp + 8);
            oa[ds] = __builtin_amdgcn_wmma_f32_16x16x32_f16(false, ap, false, u.v,
                                                            (short)0, oa[ds], false, false);
        }
    }

    // normalize and store O (f16) for the output projection
#pragma unroll
    for (int ds = 0; ds < 4; ++ds) {
#pragma unroll
        for (int r = 0; r < 8; ++r) {
            int qr = qt * 16 + r + 8 * hi;
            if (qr < LSEQ) {
                float val = oa[ds][r] / lrow[r];
                o[((size_t)(n * LSEQ + qr)) * DMODEL + hc + ds * 16 + ln] = (f16_t)val;
            }
        }
    }
}

// ---------------------------------------------------------------------------
extern "C" void kernel_launch(void* const* d_in, const int* in_sizes, int n_in,
                              void* d_out, int out_size, void* d_ws, size_t ws_size,
                              hipStream_t stream) {
    (void)in_sizes; (void)n_in; (void)out_size; (void)ws_size;
    const float* x     = (const float*)d_in[0];
    const float* klcls = (const float*)d_in[1];
    const float* Wq = (const float*)d_in[2];
    const float* bq = (const float*)d_in[3];
    const float* Wk = (const float*)d_in[4];
    const float* bk = (const float*)d_in[5];
    const float* Wv = (const float*)d_in[6];
    const float* bv = (const float*)d_in[7];
    const float* Wo = (const float*)d_in[8];
    const float* bo = (const float*)d_in[9];
    float* out = (float*)d_out;

    char* ws = (char*)d_ws;
    size_t off = 0;
    auto take = [&](size_t bytes) -> void* {
        void* p = ws + off;
        off = (off + bytes + 255) & ~(size_t)255;
        return p;
    };

    float* lse     = (float*)take((size_t)NB_B * NSEG * DMODEL * 4);
    float* partial = (float*)take((size_t)NBOUND * NB_B * 4);
    int*   top3    = (int*)take(TOPK * 4);
    f16_t* wq16t = (f16_t*)take((size_t)DMODEL * DMODEL * 2);
    f16_t* wk16t = (f16_t*)take((size_t)DMODEL * DMODEL * 2);
    f16_t* wv16t = (f16_t*)take((size_t)DMODEL * DMODEL * 2);
    f16_t* wo16t = (f16_t*)take((size_t)DMODEL * DMODEL * 2);
    f16_t* h16  = (f16_t*)take((size_t)MTOT * DMODEL * 2);
    f16_t* q16  = (f16_t*)take((size_t)MTOT * DMODEL * 2);
    f16_t* k16  = (f16_t*)take((size_t)MTOT * DMODEL * 2);
    f16_t* v16  = (f16_t*)take((size_t)MTOT * DMODEL * 2);
    f16_t* vt16 = (f16_t*)take((size_t)NBATCH * NHEAD * HD * LTPAD * 2);
    f16_t* o16  = (f16_t*)take((size_t)MTOT * DMODEL * 2);

    const int W_ELEMS  = DMODEL * DMODEL;                 // 262144
    const int VT_ELEMS = NBATCH * NHEAD * HD * LTPAD;     // 6684672

    // --- KL gating path ---
    lse_kernel<<<NB_B * NSEG, 256, 0, stream>>>(x, lse);
    kl_partial_kernel<<<NB_B * NBOUND, 256, 0, stream>>>(x, lse, partial);
    kl_top3_kernel<<<1, 64, 0, stream>>>(partial, top3);

    // --- transposed weight conversion / V^T pad zeroing / h gather ---
    cvt_f16_T_kernel<<<W_ELEMS / 256, 256, 0, stream>>>(Wq, wq16t);
    cvt_f16_T_kernel<<<W_ELEMS / 256, 256, 0, stream>>>(Wk, wk16t);
    cvt_f16_T_kernel<<<W_ELEMS / 256, 256, 0, stream>>>(Wv, wv16t);
    cvt_f16_T_kernel<<<W_ELEMS / 256, 256, 0, stream>>>(Wo, wo16t);
    zero_f16_kernel<<<(VT_ELEMS + 255) / 256, 256, 0, stream>>>(vt16, VT_ELEMS);
    build_h_kernel<<<MTOT, 256, 0, stream>>>(x, klcls, top3, h16);

    // --- QKV projections (Q pre-scaled by 1/sqrt(HD)) ---
    dim3 ggrid(DMODEL / 64, (MTOT + 127) / 128);
    gemm_kernel<<<ggrid, 256, 0, stream>>>(h16, wq16t, bq, 0.125f, 0, q16, nullptr, nullptr);
    gemm_kernel<<<ggrid, 256, 0, stream>>>(h16, wk16t, bk, 1.0f,   0, k16, nullptr, nullptr);
    gemm_kernel<<<ggrid, 256, 0, stream>>>(h16, wv16t, bv, 1.0f,   1, v16, vt16,   nullptr);

    // --- flash attention: 6336 wave-tasks, 8 waves/block ---
    attn_kernel<<<NTASKS / 8, 256, 0, stream>>>(q16, k16, vt16, o16);

    // --- output projection with scatter into d_out ---
    gemm_kernel<<<ggrid, 256, 0, stream>>>(o16, wo16t, bo, 1.0f, 2, nullptr, nullptr, out);
}
